// Head_59055800320175
// MI455X (gfx1250) — matrix operand order repaired and checked by reading, not compiled, not run on hardware
//
#include <hip/hip_runtime.h>

#define DEVI __device__ __forceinline__

// ---------------------------------------------------------------- constants
#define BB 16
#define TT 2048
#define EE 1024
#define HS 128
#define KT 64            // keys per flash-attention inner tile

// ---------------------------------------------------------------- types
typedef __bf16 bf16_t;
typedef bf16_t v16bf __attribute__((ext_vector_type(16)));
typedef float  v8f   __attribute__((ext_vector_type(8)));
typedef int    v4i_t __attribute__((ext_vector_type(4)));

union FragBF {
    v16bf          v;
    unsigned int   u[8];
    unsigned short h[16];
};

DEVI unsigned short f2bf(float f) {
    unsigned u = __float_as_uint(f);
    unsigned r = u + 0x7FFFu + ((u >> 16) & 1u);   // round-to-nearest-even
    return (unsigned short)(r >> 16);
}
DEVI unsigned pack2bf(float a, float b) {
    return (unsigned)f2bf(a) | ((unsigned)f2bf(b) << 16);
}

// -------- CDNA5 async global->LDS path (ASYNCcnt), with safe fallback
#if defined(__gfx1250__) && defined(__has_builtin)
#if __has_builtin(__builtin_amdgcn_global_load_async_to_lds_b128)
#define HAVE_ASYNC_LDS 1
#endif
#endif
#ifndef HAVE_ASYNC_LDS
#define HAVE_ASYNC_LDS 0
#endif

#if HAVE_ASYNC_LDS
// builtin signature (from hipcc diagnostic): (v4i* gaddr, v4i* ldsaddr, imm offset, imm cpol)
#define ASYNC_CP_B128(gp, lp)                                              \
    __builtin_amdgcn_global_load_async_to_lds_b128(                        \
        (v4i_t*)(gp), (v4i_t*)(lp), 0, 0)
DEVI void wait_async_lds() {
#if __has_builtin(__builtin_amdgcn_s_wait_asynccnt)
    __builtin_amdgcn_s_wait_asynccnt(0);
#else
    asm volatile("s_wait_asynccnt 0" ::: "memory");
#endif
}
#else
DEVI void wait_async_lds() {}
#endif

// ============================================================ Phase 1: X @ W
// X: [M=B*T, E] fp32 row-major, W: [E, HS] fp32 row-major, dst: [M, HS] bf16.
// One workgroup computes a 128x128 output tile; 8 waves in a 4x2 grid
// (each wave: 2x4 WMMA tiles of 16x16), K-loop in steps of 32 (bf16 WMMA K).
__global__ __launch_bounds__(256) void proj_bf16_wmma(
    const float* __restrict__ X, const float* __restrict__ W,
    unsigned short* __restrict__ dst)
{
    __shared__ unsigned short ldsX[128 * 40];   // [row][k]  stride 40 (32 + pad 8)
    __shared__ unsigned short ldsWt[128 * 40];  // [col][k]  transposed, stride 40

    const int tid  = threadIdx.x;
    const int lane = tid & 31;
    const int w    = tid >> 5;
    const int wm   = w & 3;        // row quadrant   (wm*32)
    const int wn   = w >> 2;       // col half       (wn*64)
    const int ln   = lane & 15;
    const int h    = lane >> 4;
    const int row0 = blockIdx.x * 128;

    v8f acc[2][4];
#pragma unroll
    for (int mi = 0; mi < 2; ++mi)
#pragma unroll
        for (int ni = 0; ni < 4; ++ni)
#pragma unroll
            for (int i = 0; i < 8; ++i) acc[mi][ni][i] = 0.0f;

    for (int kb = 0; kb < EE / 32; ++kb) {
        const int k0 = kb * 32;
        __syncthreads();   // protect previous iteration's LDS reads
        // ---- X tile 128x32 fp32 -> bf16 LDS
#pragma unroll
        for (int rep = 0; rep < 4; ++rep) {
            int slot = tid + rep * 256;              // 0..1023
            int r    = slot >> 3;                    // 8 float4 per row
            int c4   = slot & 7;
            const float4 f = *(const float4*)(X + (size_t)(row0 + r) * EE + k0 + c4 * 4);
            unsigned* p = (unsigned*)&ldsX[r * 40 + c4 * 4];
            p[0] = pack2bf(f.x, f.y);
            p[1] = pack2bf(f.z, f.w);
        }
        // ---- W tile 32x128 fp32 -> transposed bf16 LDS [col][k]
#pragma unroll
        for (int rep = 0; rep < 4; ++rep) {
            int slot = tid + rep * 256;              // 0..1023
            int kr   = slot >> 5;                    // 32 float4 per row
            int c4   = slot & 31;
            const float4 f = *(const float4*)(W + (size_t)(k0 + kr) * HS + c4 * 4);
            int c = c4 * 4;
            ldsWt[(c + 0) * 40 + kr] = f2bf(f.x);
            ldsWt[(c + 1) * 40 + kr] = f2bf(f.y);
            ldsWt[(c + 2) * 40 + kr] = f2bf(f.z);
            ldsWt[(c + 3) * 40 + kr] = f2bf(f.w);
        }
        __syncthreads();

        // ---- A fragments (16x32 bf16, ISA layout: lanes0-15 K0-7/16-23, lanes16-31 K8-15/24-31)
        FragBF a[2];
#pragma unroll
        for (int mi = 0; mi < 2; ++mi) {
            int lr = wm * 32 + mi * 16 + ln;
            const unsigned* px = (const unsigned*)&ldsX[lr * 40];
#pragma unroll
            for (int j = 0; j < 4; ++j) a[mi].u[j]     = px[h * 4 + j];       // K = h*8+2j
#pragma unroll
            for (int j = 0; j < 4; ++j) a[mi].u[4 + j] = px[8 + h * 4 + j];   // K = 16+h*8+2j
        }
        // ---- B fragments (32x16 bf16: lane=col, VGPR j holds K = h*8+2j pair (j<4), 16+h*8+2(j-4) pair (j>=4))
#pragma unroll
        for (int ni = 0; ni < 4; ++ni) {
            FragBF b;
            int n = wn * 64 + ni * 16 + ln;
            const unsigned* pw = (const unsigned*)&ldsWt[n * 40];
#pragma unroll
            for (int j = 0; j < 8; ++j) b.u[j] = pw[((j & 4) << 1) + h * 4 + (j & 3)];
#pragma unroll
            for (int mi = 0; mi < 2; ++mi)
                acc[mi][ni] = __builtin_amdgcn_wmma_f32_16x16x32_bf16(
                    false, a[mi].v, false, b.v, (short)0, acc[mi][ni], false, false);
        }
    }

    // ---- epilogue: C/D layout row = i + 8*h, col = ln  -> bf16 global
#pragma unroll
    for (int mi = 0; mi < 2; ++mi)
#pragma unroll
        for (int ni = 0; ni < 4; ++ni) {
            int col = wn * 64 + ni * 16 + ln;
#pragma unroll
            for (int i = 0; i < 8; ++i) {
                int row = row0 + wm * 32 + mi * 16 + i + 8 * h;
                dst[(size_t)row * HS + col] = f2bf(acc[mi][ni][i]);
            }
        }
}

// ==================================================== Phase 2: flash attention
// Per workgroup: 128 q-rows of one batch. 8 waves, each owns 16 complete rows.
// Double-buffered 64-key tiles; K/V staged with CDNA5 async global->LDS copies
// (ASYNCcnt) so the next tile's HBM fetch overlaps WMMA + softmax of this tile.
__global__ __launch_bounds__(256) void attn_flash_wmma(
    const unsigned short* __restrict__ Q,
    const unsigned short* __restrict__ K,
    const unsigned short* __restrict__ V,
    float* __restrict__ out)
{
    // row stride 136 bf16 = 272 B = 17 x 16 B  -> b128-aligned AND bank-staggered
    __shared__ unsigned short ldsK[2][KT * 136];   // [buf][key][d]
    __shared__ unsigned short ldsV[2][KT * 136];   // [buf][key][d]
    __shared__ unsigned short ldsP[128 * 72];      // [qrow][key] stride 72 (64+8)

    const int tid  = threadIdx.x;
    const int lane = tid & 31;
    const int w    = tid >> 5;
    const int ln   = lane & 15;
    const int h    = lane >> 4;
    const int bz   = blockIdx.y;
    const int qrow0 = blockIdx.x * 128;
    const int wrow  = w * 16;                   // wave's local q-row base
    const size_t base = (size_t)bz * TT * HS;

    // ---- issue one 64x128 bf16 K tile + V tile into LDS buffer `buf`
    auto issue_tile = [&](int kt, int buf) {
        const uint4* gk = (const uint4*)(K + base + (size_t)kt * KT * HS);
        const uint4* gv = (const uint4*)(V + base + (size_t)kt * KT * HS);
        uint4* lk = (uint4*)&ldsK[buf][0];
        uint4* lv = (uint4*)&ldsV[buf][0];
#pragma unroll
        for (int rep = 0; rep < 4; ++rep) {
            int chunk = tid + rep * 256;         // 0..1023 16B-chunks per tile
            int r = chunk >> 4;                  // 16 chunks per 256B row
            int c = chunk & 15;
            int ldst = r * 17 + c;               // padded stride: 17 x 16B
#if HAVE_ASYNC_LDS
            ASYNC_CP_B128(gk + chunk, lk + ldst);
            ASYNC_CP_B128(gv + chunk, lv + ldst);
#else
            lk[ldst] = gk[chunk];
            lv[ldst] = gv[chunk];
#endif
        }
    };

    // ---- load Q A-fragments straight from global (4 k-steps of 32)
    FragBF qA[4];
    {
        const unsigned* qp =
            (const unsigned*)(Q + base + (size_t)(qrow0 + wrow + ln) * HS);
#pragma unroll
        for (int ks = 0; ks < 4; ++ks) {
            int d0 = ks * 32;
#pragma unroll
            for (int j = 0; j < 4; ++j) qA[ks].u[j]     = qp[d0 / 2 + h * 4 + j];
#pragma unroll
            for (int j = 0; j < 4; ++j) qA[ks].u[4 + j] = qp[d0 / 2 + 8 + h * 4 + j];
        }
    }

    float m[8], l[8];
    v8f o[8];
#pragma unroll
    for (int i = 0; i < 8; ++i) { m[i] = -3.0e38f; l[i] = 0.0f; }
#pragma unroll
    for (int nt = 0; nt < 8; ++nt)
#pragma unroll
        for (int i = 0; i < 8; ++i) o[nt][i] = 0.0f;

    const float sc_log2e = 0.088388347648318447f * 1.4426950408889634f; // HS^-0.5 * log2(e)

    issue_tile(0, 0);                            // prologue: prefetch tile 0

    for (int kt = 0; kt < TT / KT; ++kt) {
        const int buf = kt & 1;
        wait_async_lds();       // this wave's async copies for `buf` landed
        __syncthreads();        // everyone's copies landed; prev compute done
        if (kt + 1 < TT / KT) issue_tile(kt + 1, buf ^ 1);  // overlap next fetch

        const unsigned short* tK = ldsK[buf];
        const unsigned short* tV = ldsV[buf];

        // ---- S = Q K^T : 4 col-tiles x 4 k-steps of WMMA
        v8f s[4];
#pragma unroll
        for (int nt = 0; nt < 4; ++nt) {
#pragma unroll
            for (int i = 0; i < 8; ++i) s[nt][i] = 0.0f;
            const unsigned* pk = (const unsigned*)&tK[(nt * 16 + ln) * 136];
#pragma unroll
            for (int ks = 0; ks < 4; ++ks) {
                FragBF bk;
#pragma unroll
                for (int j = 0; j < 8; ++j) bk.u[j] = pk[ks * 16 + ((j & 4) << 1) + h * 4 + (j & 3)];
                s[nt] = __builtin_amdgcn_wmma_f32_16x16x32_bf16(
                    false, qA[ks].v, false, bk.v, (short)0, s[nt], false, false);
            }
        }

        // ---- online softmax (rows live in 16-lane halves: xor 1,2,4,8)
#pragma unroll
        for (int i = 0; i < 8; ++i) {
            float mm = fmaxf(fmaxf(s[0][i], s[1][i]), fmaxf(s[2][i], s[3][i]));
            mm = fmaxf(mm, __shfl_xor(mm, 1, 32));
            mm = fmaxf(mm, __shfl_xor(mm, 2, 32));
            mm = fmaxf(mm, __shfl_xor(mm, 4, 32));
            mm = fmaxf(mm, __shfl_xor(mm, 8, 32));
            float mnew = fmaxf(m[i], mm);
            float corr = exp2f((m[i] - mnew) * sc_log2e);
            m[i] = mnew;
            l[i] *= corr;
#pragma unroll
            for (int nt = 0; nt < 8; ++nt) o[nt][i] *= corr;
        }
#pragma unroll
        for (int nt = 0; nt < 4; ++nt)
#pragma unroll
            for (int i = 0; i < 8; ++i)
                s[nt][i] = exp2f((s[nt][i] - m[i]) * sc_log2e);
#pragma unroll
        for (int i = 0; i < 8; ++i) {
            float r = s[0][i] + s[1][i] + s[2][i] + s[3][i];
            r += __shfl_xor(r, 1, 32);
            r += __shfl_xor(r, 2, 32);
            r += __shfl_xor(r, 4, 32);
            r += __shfl_xor(r, 8, 32);
            l[i] += r;
        }

        // ---- P -> LDS (wave-private rows; LDS is in-order per wave, no barrier)
#pragma unroll
        for (int nt = 0; nt < 4; ++nt) {
            int col = nt * 16 + ln;
#pragma unroll
            for (int i = 0; i < 8; ++i)
                ldsP[(wrow + i + 8 * h) * 72 + col] = f2bf(s[nt][i]);
        }

        // ---- O += P V
        FragBF pA[2];
        {
            const unsigned* pp = (const unsigned*)&ldsP[(wrow + ln) * 72];
#pragma unroll
            for (int ks = 0; ks < 2; ++ks) {
#pragma unroll
                for (int j = 0; j < 4; ++j) pA[ks].u[j]     = pp[ks * 16 + h * 4 + j];
#pragma unroll
                for (int j = 0; j < 4; ++j) pA[ks].u[4 + j] = pp[ks * 16 + 8 + h * 4 + j];
            }
        }
#pragma unroll
        for (int nt = 0; nt < 8; ++nt) {
            int n = nt * 16 + ln;
#pragma unroll
            for (int ks = 0; ks < 2; ++ks) {
                FragBF bv;
#pragma unroll
                for (int j = 0; j < 8; ++j) {
                    int k = ks * 32 + ((j & 4) << 2) + h * 8 + 2 * (j & 3);
                    bv.h[2 * j]     = tV[k * 136 + n];
                    bv.h[2 * j + 1] = tV[(k + 1) * 136 + n];
                }
                o[nt] = __builtin_amdgcn_wmma_f32_16x16x32_bf16(
                    false, pA[ks].v, false, bv.v, (short)0, o[nt], false, false);
            }
        }
    }

    // ---- normalize + store fp32
#pragma unroll
    for (int i = 0; i < 8; ++i) l[i] = 1.0f / l[i];
#pragma unroll
    for (int nt = 0; nt < 8; ++nt) {
        int col = nt * 16 + ln;
#pragma unroll
        for (int i = 0; i < 8; ++i) {
            int row = qrow0 + wrow + i + 8 * h;
            out[base + (size_t)row * HS + col] = o[nt][i] * l[i];
        }
    }
}

// ---------------------------------------------------------------- launcher
extern "C" void kernel_launch(void* const* d_in, const int* in_sizes, int n_in,
                              void* d_out, int out_size, void* d_ws, size_t ws_size,
                              hipStream_t stream) {
    const float* x_head = (const float*)d_in[0];
    const float* x_body = (const float*)d_in[1];
    const float* Wk     = (const float*)d_in[2];
    const float* Wq     = (const float*)d_in[3];
    const float* Wv     = (const float*)d_in[4];
    float* out          = (float*)d_out;

    const size_t mat = (size_t)BB * TT * HS;           // 4,194,304 elements
    unsigned short* ws = (unsigned short*)d_ws;        // 6 * mat bf16 = 48 MB

    for (int s = 0; s < 2; ++s) {
        const float* x = s ? x_body : x_head;
        unsigned short* q = ws + (size_t)s * 3 * mat;
        unsigned short* k = q + mat;
        unsigned short* v = k + mat;
        proj_bf16_wmma<<<dim3((BB * TT) / 128), 256, 0, stream>>>(x, Wq, q);
        proj_bf16_wmma<<<dim3((BB * TT) / 128), 256, 0, stream>>>(x, Wk, k);
        proj_bf16_wmma<<<dim3((BB * TT) / 128), 256, 0, stream>>>(x, Wv, v);
    }
    for (int s = 0; s < 2; ++s) {
        unsigned short* q = ws + (size_t)s * 3 * mat;
        unsigned short* k = q + mat;
        unsigned short* v = k + mat;
        attn_flash_wmma<<<dim3(TT / 128, BB), 256, 0, stream>>>(q, k, v, out + s * mat);
    }
}
